// FFNMoE_63513976373306
// MI455X (gfx1250) — matrix operand order repaired and checked by reading, not compiled, not run on hardware
//
#include <hip/hip_runtime.h>
#include <math.h>

// Problem constants (match reference)
#define DMODEL 1024
#define NEXP   8
#define TOPK   2
#define DFF    4096
#define NTOK   4096          // B*S = 2*2048
#define EPS_LN 1e-5f
#define EPS_TK 1e-8f

typedef __attribute__((ext_vector_type(16))) __bf16 v16bf;
typedef __attribute__((ext_vector_type(8)))  float  v8f;

// issue one async 16B memory->LDS copy (gfx1250 ASYNCcnt path)
__device__ __forceinline__ void async_copy_b128(void* lds_dst, const void* gsrc) {
  const unsigned lds = (unsigned)(uintptr_t)lds_dst;               // low 32 bits = DS offset
  const unsigned long long g = (unsigned long long)(uintptr_t)gsrc;
  asm volatile("global_load_async_to_lds_b128 %0, %1, off"
               :: "v"(lds), "v"(g) : "memory");
}
__device__ __forceinline__ void async_wait0() {
  asm volatile("s_wait_asynccnt 0x0" ::: "memory");
}

// ---------------------------------------------------------------------------
// Zero per-expert dispatch counters (graph-capture safe re-init each launch)
// ---------------------------------------------------------------------------
__global__ __launch_bounds__(32) void zero_counts_kernel(int* counts) {
  if (threadIdx.x < NEXP) counts[threadIdx.x] = 0;
}

// ---------------------------------------------------------------------------
// LayerNorm + router (softmax/top-2) + dispatch-list build + residual init
// ---------------------------------------------------------------------------
__global__ __launch_bounds__(128) void ln_gate_kernel(
    const float* __restrict__ x, const float* __restrict__ gamma,
    const float* __restrict__ beta, const float* __restrict__ Wg,
    const float* __restrict__ bg,
    __bf16* __restrict__ xn, float* __restrict__ out,
    int* __restrict__ counts, int* __restrict__ idx_tok,
    int* __restrict__ idx_hrow, float* __restrict__ wts)
{
  const int t    = blockIdx.x;
  const int tid  = threadIdx.x;
  const int wave = tid >> 5;
  const int lane = tid & 31;
  const float* xr = x + (size_t)t * DMODEL;

  float v[8];
  float s = 0.f;
#pragma unroll
  for (int i = 0; i < 8; ++i) { v[i] = xr[tid + i * 128]; s += v[i]; }

  __shared__ float red[4];
  __shared__ float bc;
#pragma unroll
  for (int o = 16; o; o >>= 1) s += __shfl_xor(s, o, 32);
  if (!lane) red[wave] = s;
  __syncthreads();
  if (!tid) bc = (red[0] + red[1] + red[2] + red[3]) * (1.f / DMODEL);
  __syncthreads();
  const float mean = bc;

  float vs = 0.f;
#pragma unroll
  for (int i = 0; i < 8; ++i) { float d = v[i] - mean; vs += d * d; }
#pragma unroll
  for (int o = 16; o; o >>= 1) vs += __shfl_xor(vs, o, 32);
  if (!lane) red[wave] = vs;
  __syncthreads();
  if (!tid) bc = rsqrtf((red[0] + red[1] + red[2] + red[3]) * (1.f / DMODEL) + EPS_LN);
  __syncthreads();
  const float rstd = bc;

  float lg[NEXP];
#pragma unroll
  for (int e = 0; e < NEXP; ++e) lg[e] = 0.f;
#pragma unroll
  for (int i = 0; i < 8; ++i) {
    const int d = tid + i * 128;
    const float h = (v[i] - mean) * rstd * gamma[d] + beta[d];
    xn[(size_t)t * DMODEL + d]  = (__bf16)h;
    out[(size_t)t * DMODEL + d] = v[i];          // residual base: out = x
#pragma unroll
    for (int e = 0; e < NEXP; ++e) lg[e] += h * Wg[e * DMODEL + d];
  }

  __shared__ float lred[4][NEXP];
#pragma unroll
  for (int e = 0; e < NEXP; ++e) {
    float q = lg[e];
#pragma unroll
    for (int o = 16; o; o >>= 1) q += __shfl_xor(q, o, 32);
    if (!lane) lred[wave][e] = q;
  }
  __syncthreads();

  if (!tid) {
    float p[NEXP];
    float mx = -1e30f;
    for (int e = 0; e < NEXP; ++e) {
      p[e] = lred[0][e] + lred[1][e] + lred[2][e] + lred[3][e] + bg[e];
      mx = fmaxf(mx, p[e]);
    }
    float sum = 0.f;
    for (int e = 0; e < NEXP; ++e) { p[e] = __expf(p[e] - mx); sum += p[e]; }
    const float inv = 1.f / sum;
    for (int e = 0; e < NEXP; ++e) p[e] *= inv;
    int i0 = 0;
    for (int e = 1; e < NEXP; ++e) if (p[e] > p[i0]) i0 = e;
    int i1 = (i0 == 0) ? 1 : 0;
    for (int e = 0; e < NEXP; ++e) if (e != i0 && p[e] > p[i1]) i1 = e;
    const float denom = p[i0] + p[i1] + EPS_TK;
    const float w0 = p[i0] / denom;
    const float w1 = p[i1] / denom;

    int pos = atomicAdd(&counts[i0], 1);
    idx_tok [i0 * NTOK + pos] = t;
    idx_hrow[i0 * NTOK + pos] = t * TOPK + 0;
    wts     [i0 * NTOK + pos] = w0;
    pos = atomicAdd(&counts[i1], 1);
    idx_tok [i1 * NTOK + pos] = t;
    idx_hrow[i1 * NTOK + pos] = t * TOPK + 1;
    wts     [i1 * NTOK + pos] = w1;
  }
}

// ---------------------------------------------------------------------------
// Grouped GEMM 1: H[hrow, f] = gelu( gather(xn) @ W1[e]^T + b1[e] )
// block tile 64(M) x 64(N), K-step 32; 4 waves, each 16x64 via 4 wmma accs
// A tile staged via global_load_async_to_lds_b128 (ASYNCcnt), B tile via
// float4 loads + cvt_pk_bf16 + ds_store_b128.
// ---------------------------------------------------------------------------
__global__ __launch_bounds__(128) void gemm1_kernel(
    const __bf16* __restrict__ xn, const float* __restrict__ W1,
    const float* __restrict__ b1,
    const int* __restrict__ counts, const int* __restrict__ idx_tok,
    const int* __restrict__ idx_hrow,
    __bf16* __restrict__ H)
{
  const int e  = blockIdx.z;
  const int M  = counts[e];
  const int m0 = blockIdx.y * 64;
  if (m0 >= M) return;                    // grouped-GEMM early exit
  const int n0 = blockIdx.x * 64;

  const int tid  = threadIdx.x;
  const int wave = tid >> 5;
  const int lane = tid & 31;
  const int hf   = lane >> 4;             // lane half (ISA 16-bit frag layout)
  const int l16  = lane & 15;

  __shared__ __attribute__((aligned(16))) __bf16 As[64][40];
  __shared__ __attribute__((aligned(16))) __bf16 Bs[64][40];
  __shared__ int toks[64];
  __shared__ int hrows[64];

  if (tid < 64) {
    const int m  = m0 + tid;
    const int mm = (m < M) ? m : m0;      // duplicate a valid row for tail
    toks[tid]  = idx_tok [e * NTOK + mm];
    hrows[tid] = idx_hrow[e * NTOK + mm];
  }
  __syncthreads();

  const float* W1e = W1 + (size_t)e * DFF * DMODEL;

  // each thread owns 2 (row, 8-elem chunk) slots of the 64x32 tiles
  const int ch = tid & 3;                 // chunk: cols [8*ch, 8*ch+8)
  const int r0 = tid >> 2;                // rows r0 and r0+32
  const __bf16* ga[2];
  const float*  gb[2];
  void* lA[2];
  void* lB[2];
#pragma unroll
  for (int i = 0; i < 2; ++i) {
    const int r = r0 + 32 * i;
    ga[i] = xn + (size_t)toks[r] * DMODEL + ch * 8;
    gb[i] = W1e + (size_t)(n0 + r) * DMODEL + ch * 8;
    lA[i] = (void*)&As[r][ch * 8];
    lB[i] = (void*)&Bs[r][ch * 8];
  }

  v8f acc[4] = {};

  for (int k0 = 0; k0 < DMODEL; k0 += 32) {
#pragma unroll
    for (int i = 0; i < 2; ++i) {
      async_copy_b128(lA[i], ga[i] + k0);             // bf16 A tile: direct to LDS
      const float4 f0 = *(const float4*)(gb[i] + k0);
      const float4 f1 = *(const float4*)(gb[i] + k0 + 4);
      union { uint4 u; __bf16 h[8]; } pk;
      pk.h[0] = (__bf16)f0.x; pk.h[1] = (__bf16)f0.y;
      pk.h[2] = (__bf16)f0.z; pk.h[3] = (__bf16)f0.w;
      pk.h[4] = (__bf16)f1.x; pk.h[5] = (__bf16)f1.y;
      pk.h[6] = (__bf16)f1.z; pk.h[7] = (__bf16)f1.w;
      *(uint4*)lB[i] = pk.u;
    }
    if (k0 + 32 < DMODEL)
      __builtin_prefetch(gb[0] + k0 + 32, 0, 1);      // global_prefetch_b8
    async_wait0();
    __syncthreads();

    // A fragment: 16x32 bf16 (ISA 7.12.2 lane layout)
    v16bf a;
    const int arow = wave * 16 + l16;
#pragma unroll
    for (int vv = 0; vv < 8; ++vv)
#pragma unroll
      for (int ss = 0; ss < 2; ++ss)
        a[2 * vv + ss] = As[arow][((vv < 4) ? 0 : 16) + hf * 8 + (vv & 3) * 2 + ss];

#pragma unroll
    for (int nt = 0; nt < 4; ++nt) {
      v16bf b;
      const int bcol = nt * 16 + l16;
#pragma unroll
      for (int vv = 0; vv < 8; ++vv)
#pragma unroll
        for (int ss = 0; ss < 2; ++ss)
          b[2 * vv + ss] = Bs[bcol][((vv < 4) ? 0 : 16) + hf * 8 + (vv & 3) * 2 + ss];

      acc[nt] = __builtin_amdgcn_wmma_f32_16x16x32_bf16(
          false, a, false, b, (short)0, acc[nt], false, false);
    }
    __syncthreads();
  }

  // epilogue: bias + exact gelu -> bf16 hidden rows
  const float* b1e = b1 + (size_t)e * DFF;
#pragma unroll
  for (int nt = 0; nt < 4; ++nt)
#pragma unroll
    for (int vv = 0; vv < 8; ++vv) {
      const int mloc = wave * 16 + vv + 8 * hf;   // C-layout: VGPR v, halves
      if (m0 + mloc < M) {
        const int f = n0 + nt * 16 + l16;
        const float z = acc[nt][vv] + b1e[f];
        const float g = 0.5f * z * (1.f + erff(z * 0.70710678118654752f));
        H[(size_t)hrows[mloc] * DFF + f] = (__bf16)g;
      }
    }
}

// ---------------------------------------------------------------------------
// Grouped GEMM 2: out[tok] += w * ( H @ W2[e]^T + b2[e] )  (atomic combine)
// ---------------------------------------------------------------------------
__global__ __launch_bounds__(128) void gemm2_kernel(
    const __bf16* __restrict__ H, const float* __restrict__ W2,
    const float* __restrict__ b2,
    const int* __restrict__ counts, const int* __restrict__ idx_tok,
    const int* __restrict__ idx_hrow, const float* __restrict__ wts,
    float* __restrict__ out)
{
  const int e  = blockIdx.z;
  const int M  = counts[e];
  const int m0 = blockIdx.y * 64;
  if (m0 >= M) return;
  const int n0 = blockIdx.x * 64;

  const int tid  = threadIdx.x;
  const int wave = tid >> 5;
  const int lane = tid & 31;
  const int hf   = lane >> 4;
  const int l16  = lane & 15;

  __shared__ __attribute__((aligned(16))) __bf16 As[64][40];
  __shared__ __attribute__((aligned(16))) __bf16 Bs[64][40];
  __shared__ int   toks[64];
  __shared__ int   hrows[64];
  __shared__ float wvals[64];

  if (tid < 64) {
    const int m  = m0 + tid;
    const int mm = (m < M) ? m : m0;
    toks[tid]  = idx_tok [e * NTOK + mm];
    hrows[tid] = idx_hrow[e * NTOK + mm];
    wvals[tid] = wts     [e * NTOK + mm];
  }
  __syncthreads();

  const float* W2e = W2 + (size_t)e * DMODEL * DFF;

  const int ch = tid & 3;
  const int r0 = tid >> 2;
  const __bf16* ga[2];
  const float*  gb[2];
  void* lA[2];
  void* lB[2];
#pragma unroll
  for (int i = 0; i < 2; ++i) {
    const int r = r0 + 32 * i;
    ga[i] = H + (size_t)hrows[r] * DFF + ch * 8;
    gb[i] = W2e + (size_t)(n0 + r) * DFF + ch * 8;
    lA[i] = (void*)&As[r][ch * 8];
    lB[i] = (void*)&Bs[r][ch * 8];
  }

  v8f acc[4] = {};

  for (int k0 = 0; k0 < DFF; k0 += 32) {
#pragma unroll
    for (int i = 0; i < 2; ++i) {
      async_copy_b128(lA[i], ga[i] + k0);
      const float4 f0 = *(const float4*)(gb[i] + k0);
      const float4 f1 = *(const float4*)(gb[i] + k0 + 4);
      union { uint4 u; __bf16 h[8]; } pk;
      pk.h[0] = (__bf16)f0.x; pk.h[1] = (__bf16)f0.y;
      pk.h[2] = (__bf16)f0.z; pk.h[3] = (__bf16)f0.w;
      pk.h[4] = (__bf16)f1.x; pk.h[5] = (__bf16)f1.y;
      pk.h[6] = (__bf16)f1.z; pk.h[7] = (__bf16)f1.w;
      *(uint4*)lB[i] = pk.u;
    }
    if (k0 + 32 < DFF)
      __builtin_prefetch(gb[0] + k0 + 32, 0, 1);
    async_wait0();
    __syncthreads();

    v16bf a;
    const int arow = wave * 16 + l16;
#pragma unroll
    for (int vv = 0; vv < 8; ++vv)
#pragma unroll
      for (int ss = 0; ss < 2; ++ss)
        a[2 * vv + ss] = As[arow][((vv < 4) ? 0 : 16) + hf * 8 + (vv & 3) * 2 + ss];

#pragma unroll
    for (int nt = 0; nt < 4; ++nt) {
      v16bf b;
      const int bcol = nt * 16 + l16;
#pragma unroll
      for (int vv = 0; vv < 8; ++vv)
#pragma unroll
        for (int ss = 0; ss < 2; ++ss)
          b[2 * vv + ss] = Bs[bcol][((vv < 4) ? 0 : 16) + hf * 8 + (vv & 3) * 2 + ss];

      acc[nt] = __builtin_amdgcn_wmma_f32_16x16x32_bf16(
          false, a, false, b, (short)0, acc[nt], false, false);
    }
    __syncthreads();
  }

  // epilogue: bias + weighted scatter-accumulate into residual output
  const float* b2e = b2 + (size_t)e * DMODEL;
#pragma unroll
  for (int nt = 0; nt < 4; ++nt)
#pragma unroll
    for (int vv = 0; vv < 8; ++vv) {
      const int mloc = wave * 16 + vv + 8 * hf;
      if (m0 + mloc < M) {
        const int col = n0 + nt * 16 + l16;
        const float z = acc[nt][vv] + b2e[col];
        atomicAdd(&out[(size_t)toks[mloc] * DMODEL + col], wvals[mloc] * z);
      }
    }
}

// ---------------------------------------------------------------------------
// launch
// ---------------------------------------------------------------------------
extern "C" void kernel_launch(void* const* d_in, const int* in_sizes, int n_in,
                              void* d_out, int out_size, void* d_ws, size_t ws_size,
                              hipStream_t stream)
{
  (void)in_sizes; (void)n_in; (void)out_size; (void)ws_size;
  const float* x     = (const float*)d_in[0];
  const float* gamma = (const float*)d_in[1];
  const float* beta  = (const float*)d_in[2];
  const float* Wg    = (const float*)d_in[3];
  const float* bg    = (const float*)d_in[4];
  const float* W1    = (const float*)d_in[5];
  const float* b1    = (const float*)d_in[6];
  const float* W2    = (const float*)d_in[7];
  const float* b2    = (const float*)d_in[8];
  float* out = (float*)d_out;

  char* ws = (char*)d_ws;
  size_t off = 0;
  __bf16* xn = (__bf16*)(ws + off); off += (size_t)NTOK * DMODEL * sizeof(__bf16);
  __bf16* H  = (__bf16*)(ws + off); off += (size_t)NTOK * TOPK * DFF * sizeof(__bf16);
  int*   counts   = (int*)(ws + off);   off += 256;
  int*   idx_tok  = (int*)(ws + off);   off += (size_t)NEXP * NTOK * sizeof(int);
  int*   idx_hrow = (int*)(ws + off);   off += (size_t)NEXP * NTOK * sizeof(int);
  float* wts      = (float*)(ws + off); off += (size_t)NEXP * NTOK * sizeof(float);

  zero_counts_kernel<<<1, 32, 0, stream>>>(counts);
  ln_gate_kernel<<<NTOK, 128, 0, stream>>>(x, gamma, beta, Wg, bg,
                                           xn, out, counts, idx_tok, idx_hrow, wts);
  gemm1_kernel<<<dim3(DFF / 64, NTOK / 64, NEXP), 128, 0, stream>>>(
      xn, W1, b1, counts, idx_tok, idx_hrow, H);
  gemm2_kernel<<<dim3(DMODEL / 64, NTOK / 64, NEXP), 128, 0, stream>>>(
      H, W2, b2, counts, idx_tok, idx_hrow, wts, out);
}
